// CausalAttNet_15066745274955
// MI455X (gfx1250) — matrix operand order repaired and checked by reading, not compiled, not run on hardware
//
#include <hip/hip_runtime.h>
#include <hip/hip_bf16.h>
#include <stdint.h>

// ---------------------------------------------------------------------------
// CausalAttNet edge top-k pooling for MI455X (gfx1250, wave32, WMMA + TDM).
//
// Roofline: ~600MB mandatory traffic @23.3TB/s ~= 26us; compute ~0.1 GFLOP ->
// bandwidth bound. GEMV runs on v_wmma_f32_16x16x32_f16 (both weight columns
// in one B matrix). W is staged to LDS via the Tensor Data Mover; the big
// x-row permutation copy is done with async global<->LDS DMA (ASYNCcnt).
// ---------------------------------------------------------------------------

#define N_NODES   100000
#define N_EDGES   3200000
#define DIM       256
#define N_GRAPHS  500
#define EPG       6400        // edges per graph (fixed by construction)
#define K_KEEP    5120        // ceil(0.8 * 6400)

typedef __attribute__((ext_vector_type(16))) _Float16 v16h;
typedef __attribute__((ext_vector_type(8)))  float    v8f;
typedef __attribute__((ext_vector_type(4)))  unsigned v4u;
typedef __attribute__((ext_vector_type(8)))  unsigned v8u;

// ---------------------------------------------------------------------------
// Kernel 1: s_row[n] = x[n,:]·W[0:256], s_col[n] = x[n,:]·W[256:512] via WMMA.
// One wave (32 threads) per 16-node tile (100000/16 = 6250 tiles exactly).
// W (2KB) is DMA'd into LDS once per block by the TDM (tensor_load_to_lds,
// 1x512 tile of 4B elements), waited on with s_wait_tensorcnt.
// A layout (16-bit 16x32, ISA 7.12.2): lane m=l&15 -> row M=m; g=l>>4 selects
// K=16g+j, j=0..15 (code's contiguous order, same as B). B: lane -> column N=l&15, half j ->
// K=16g+j. D (f32 16x16): lane N=l&15, acc[r] = D[M=8g+r, N].
// ---------------------------------------------------------------------------
__global__ __launch_bounds__(32)
void score_wmma_kernel(const float* __restrict__ x,
                       const float* __restrict__ W,   // 512 floats
                       float* __restrict__ s_row,
                       float* __restrict__ s_col) {
  __shared__ __align__(16) float wsh[2 * DIM];

  const int base = blockIdx.x * 16;
  const int lane = threadIdx.x;
  const int m = lane & 15;
  const int g = lane >> 4;
  const int row = base + m;

  // ---- TDM: DMA all 512 W floats into LDS (D# per ISA cdna5 ch.8) --------
  {
    unsigned long long ga = (unsigned long long)(const void*)W;
    unsigned lds = (unsigned)(unsigned long long)(void*)wsh;
    v4u g0 = { 1u,                                  // count=1 user descriptor
               lds,                                 // lds_addr
               (unsigned)ga,                        // global_addr[31:0]
               ((unsigned)(ga >> 32) & 0x01FFFFFFu) // global_addr[56:32]
                 | 0x80000000u };                   // type=2 ("image")
    v8u g1 = { 0x00020000u,            // wg_mask=0, data_size=2 (4 bytes)
               (512u & 0xFFFFu) << 16, // tensor_dim0[15:0] = 512
               0x00010000u,            // tensor_dim0[31:16]=0, tensor_dim1=1
               512u << 16,             // tensor_dim1 hi=0, tile_dim0=512
               1u,                     // tile_dim1=1, tile_dim2=0
               512u,                   // tensor_dim0_stride lo = 512
               0u,                     // stride0 hi=0, tensor_dim1_stride lo=0
               0u };
    asm volatile("tensor_load_to_lds %0, %1" :: "s"(g0), "s"(g1) : "memory");
    __builtin_amdgcn_s_wait_tensorcnt(0);   // TENSORcnt == 0 -> W in LDS
  }

  v8f acc = {};
  const float* xrow = x + (size_t)row * DIM;

  #pragma unroll
  for (int kc = 0; kc < DIM; kc += 32) {
    // ---- A tile: 16 nodes x 32 features, converted f32 -> f16
    v16h A;
    const float* xp = xrow + kc + 16 * g;
    #pragma unroll
    for (int j = 0; j < 8; ++j) A[j]     = (_Float16)xp[j];
    #pragma unroll
    for (int j = 0; j < 8; ++j) A[8 + j] = (_Float16)xp[8 + j];

    // ---- B tile from LDS: col 0 = W[:256] chunk, col 1 = W[256:], rest 0
    v16h B;
    if (m < 2) {
      const float* wp = wsh + m * DIM + kc + 16 * g;
      #pragma unroll
      for (int j = 0; j < 16; ++j) B[j] = (_Float16)wp[j];
    } else {
      #pragma unroll
      for (int j = 0; j < 16; ++j) B[j] = (_Float16)0.0f;
    }

    acc = __builtin_amdgcn_wmma_f32_16x16x32_f16(
        /*neg_a=*/false, A, /*neg_b=*/false, B,
        /*c_mod=*/(short)0, acc, /*reuse_a=*/false, /*reuse_b=*/false);
  }

  // Readout: column N==0 -> s_row, N==1 -> s_col.
  if (m == 0) {
    float* out = s_row + base + 8 * g;
    #pragma unroll
    for (int r = 0; r < 8; ++r) out[r] = acc[r];
  } else if (m == 1) {
    float* out = s_col + base + 8 * g;
    #pragma unroll
    for (int r = 0; r < 8; ++r) out[r] = acc[r];
  }
}

// ---------------------------------------------------------------------------
// Kernel 2: escore[e] = s_row[row[e]] + s_col[col[e]] + b
// ---------------------------------------------------------------------------
__global__ void edge_score_kernel(const float* __restrict__ s_row,
                                  const float* __restrict__ s_col,
                                  const int* __restrict__ ei,   // [2,E] flat
                                  const float* __restrict__ b,
                                  float* __restrict__ escore) {
  int e = blockIdx.x * blockDim.x + threadIdx.x;
  if (e >= N_EDGES) return;
  // gfx1250 speculative prefetch of upcoming index data (global_prefetch_b8)
  __builtin_prefetch(ei + e + 8192, 0, 0);
  __builtin_prefetch(ei + N_EDGES + e + 8192, 0, 0);
  int r = ei[e];
  int c = ei[N_EDGES + e];
  escore[e] = s_row[r] + s_col[c] + b[0];
}

// ---------------------------------------------------------------------------
// Kernel 3: zero-init selection flags + tie counters
// ---------------------------------------------------------------------------
__global__ void init_kernel(int* selc, int* self_, int* tiecnt) {
  int i = blockIdx.x * blockDim.x + threadIdx.x;
  if (i < N_NODES) { selc[i] = 0; self_[i] = 0; }
  if (i < N_GRAPHS) tiecnt[i] = 0;
}

// monotone float -> uint key (larger float => larger uint)
__device__ __forceinline__ unsigned fkey(float f) {
  unsigned bits = __float_as_uint(f);
  return bits ^ ((((int)bits) >> 31) ? 0xFFFFFFFFu : 0x80000000u);
}

// ---------------------------------------------------------------------------
// Kernel 4: per-graph radix select of the K_KEEP-th largest score.
// One 256-thread block per graph; 4 MSB->LSB digit passes over 6400 keys.
// ---------------------------------------------------------------------------
__global__ __launch_bounds__(256)
void topk_thresh_kernel(const float* __restrict__ escore,
                        unsigned* __restrict__ thr,
                        int* __restrict__ quota) {
  __shared__ unsigned hist[256];
  __shared__ unsigned s_prefix;
  __shared__ int s_remaining;
  const int g = blockIdx.x;
  const int t = threadIdx.x;
  const float* sc = escore + (size_t)g * EPG;

  if (t == 0) { s_prefix = 0u; s_remaining = K_KEEP; }
  __syncthreads();

  for (int shift = 24; shift >= 0; shift -= 8) {
    hist[t] = 0u;
    __syncthreads();
    const unsigned prefix = s_prefix;
    const unsigned himask = (shift == 24) ? 0u : (0xFFFFFFFFu << (shift + 8));
    for (int e = t; e < EPG; e += 256) {
      unsigned u = fkey(sc[e]);
      if ((u & himask) == prefix)
        atomicAdd(&hist[(u >> shift) & 255], 1u);
    }
    __syncthreads();
    if (t == 0) {
      int rem = s_remaining;
      unsigned acc = 0;
      int d = 255;
      for (; d >= 0; --d) {
        if (acc + hist[d] >= (unsigned)rem) break;
        acc += hist[d];
      }
      if (d < 0) d = 0;
      s_prefix = prefix | ((unsigned)d << shift);
      s_remaining = rem - (int)acc;
    }
    __syncthreads();
  }
  if (t == 0) { thr[g] = s_prefix; quota[g] = s_remaining; }
}

// ---------------------------------------------------------------------------
// Kernel 5: build mask, causal/conf edge weights, mark incident nodes.
// ---------------------------------------------------------------------------
__global__ void mask_kernel(const float* __restrict__ escore,
                            const unsigned* __restrict__ thr,
                            const int* __restrict__ quota,
                            int* __restrict__ tiecnt,
                            const int* __restrict__ ei,
                            int* __restrict__ selc,
                            int* __restrict__ self_,
                            float* __restrict__ maskf,
                            float* __restrict__ cw,
                            float* __restrict__ fw) {
  int e = blockIdx.x * blockDim.x + threadIdx.x;
  if (e >= N_EDGES) return;
  float s = escore[e];
  unsigned u = fkey(s);
  int g = e / EPG;
  unsigned T = thr[g];
  int keep;
  if (u > T)       keep = 1;
  else if (u == T) keep = (atomicAdd(&tiecnt[g], 1) < quota[g]) ? 1 : 0;
  else             keep = 0;

  maskf[e] = (float)keep;
  cw[e] = keep ? s : 0.0f;
  fw[e] = keep ? 0.0f : -s;

  int r = ei[e];
  int c = ei[N_EDGES + e];
  if (keep) { selc[r] = 1; selc[c] = 1; }
  else      { self_[r] = 1; self_[c] = 1; }
}

// ---------------------------------------------------------------------------
// Kernels 6-8: two-level inclusive scan of selc/self_ (1024 elems per block).
// ---------------------------------------------------------------------------
#define SCAN_T 256
#define SCAN_ELEMS 1024
#define SCAN_BLOCKS ((N_NODES + SCAN_ELEMS - 1) / SCAN_ELEMS)   // 98

__global__ __launch_bounds__(SCAN_T)
void scan1_kernel(const int* __restrict__ selc, const int* __restrict__ self_,
                  int* __restrict__ cumc, int* __restrict__ cumf,
                  int* __restrict__ bsums) {
  __shared__ int sh[SCAN_T];
  const int blk = blockIdx.x;
  const int t = threadIdx.x;
  const int base = blk * SCAN_ELEMS + t * 4;

  for (int pass = 0; pass < 2; ++pass) {
    const int* sel = pass ? self_ : selc;
    int* cum = pass ? cumf : cumc;
    int v[4]; int s = 0;
    #pragma unroll
    for (int i = 0; i < 4; ++i) {
      int idx = base + i;
      v[i] = (idx < N_NODES) ? sel[idx] : 0;
      s += v[i];
    }
    sh[t] = s;
    __syncthreads();
    for (int off = 1; off < SCAN_T; off <<= 1) {
      int xv = (t >= off) ? sh[t - off] : 0;
      __syncthreads();
      sh[t] += xv;
      __syncthreads();
    }
    int run = (t == 0) ? 0 : sh[t - 1];
    #pragma unroll
    for (int i = 0; i < 4; ++i) {
      run += v[i];
      int idx = base + i;
      if (idx < N_NODES) cum[idx] = run;
    }
    if (t == SCAN_T - 1) bsums[pass * SCAN_BLOCKS + blk] = sh[SCAN_T - 1];
    __syncthreads();
  }
}

__global__ __launch_bounds__(128)
void scan2_kernel(int* __restrict__ bsums, int* __restrict__ totals,
                  float* __restrict__ n_causal_out,
                  float* __restrict__ n_conf_out) {
  __shared__ int sh[128];
  const int t = threadIdx.x;
  for (int pass = 0; pass < 2; ++pass) {
    int v = (t < SCAN_BLOCKS) ? bsums[pass * SCAN_BLOCKS + t] : 0;
    sh[t] = v;
    __syncthreads();
    for (int off = 1; off < 128; off <<= 1) {
      int xv = (t >= off) ? sh[t - off] : 0;
      __syncthreads();
      sh[t] += xv;
      __syncthreads();
    }
    if (t < SCAN_BLOCKS) bsums[pass * SCAN_BLOCKS + t] = (t == 0) ? 0 : sh[t - 1];
    if (t == SCAN_BLOCKS - 1) {
      totals[pass] = sh[t];
      if (pass == 0) *n_causal_out = (float)sh[t];
      else           *n_conf_out   = (float)sh[t];
    }
    __syncthreads();
  }
}

__global__ void scan3_kernel(int* __restrict__ cumc, int* __restrict__ cumf,
                             const int* __restrict__ bsums) {
  int i = blockIdx.x * blockDim.x + threadIdx.x;
  if (i >= N_NODES) return;
  int blk = i / SCAN_ELEMS;
  cumc[i] += bsums[blk];
  cumf[i] += bsums[SCAN_BLOCKS + blk];
}

// ---------------------------------------------------------------------------
// Kernel 9: scatter x rows into the two compacted orders (argsort(~sel)).
// One 64-thread block per node. The 1KB row is staged global->LDS with
// GLOBAL_LOAD_ASYNC_TO_LDS_B128 (ASYNCcnt) and drained to both destination
// rows with GLOBAL_STORE_ASYNC_FROM_LDS_B128 - no VGPR round trip.
// ---------------------------------------------------------------------------
__global__ __launch_bounds__(64)
void scatter_x_kernel(const float* __restrict__ x, const int* __restrict__ batch,
                      const int* __restrict__ selc, const int* __restrict__ self_,
                      const int* __restrict__ cumc, const int* __restrict__ cumf,
                      const int* __restrict__ totals,
                      float* __restrict__ cx, float* __restrict__ cb,
                      float* __restrict__ fx, float* __restrict__ fb) {
  __shared__ __align__(16) float srow[DIM];   // 1KB staging buffer
  const int n = blockIdx.x;
  const int t = threadIdx.x;
  const int sc = selc[n], sf = self_[n];
  const int csc = cumc[n], csf = cumf[n];
  const int pc = sc ? (csc - 1) : (totals[0] + n - csc);
  const int pf = sf ? (csf - 1) : (totals[1] + n - csf);

  unsigned lds = (unsigned)(unsigned long long)(void*)(srow + t * 4);

  // global -> LDS (64 lanes x 16B = full row)
  {
    unsigned long long ga =
        (unsigned long long)(const void*)(x + (size_t)n * DIM + t * 4);
    asm volatile("global_load_async_to_lds_b128 %0, %1, off"
                 :: "v"(lds), "v"(ga) : "memory");
    asm volatile("s_wait_asynccnt 0x0" ::: "memory");
  }
  __syncthreads();

  // LDS -> both permuted destination rows
  {
    unsigned long long gc =
        (unsigned long long)(void*)(cx + (size_t)pc * DIM + t * 4);
    unsigned long long gf =
        (unsigned long long)(void*)(fx + (size_t)pf * DIM + t * 4);
    asm volatile("global_store_async_from_lds_b128 %0, %1, off"
                 :: "v"(gc), "v"(lds) : "memory");
    asm volatile("global_store_async_from_lds_b128 %0, %1, off"
                 :: "v"(gf), "v"(lds) : "memory");
    asm volatile("s_wait_asynccnt 0x0" ::: "memory");
  }

  if (t == 0) {
    float bv = (float)batch[n];
    cb[pc] = sc ? bv : -1.0f;
    fb[pf] = sf ? bv : -1.0f;
  }
}

// ---------------------------------------------------------------------------
// Kernel 10: relabel edges (new node ids, -1 padding for dropped edges).
// ---------------------------------------------------------------------------
__global__ void relabel_kernel(const int* __restrict__ ei,
                               const float* __restrict__ maskf,
                               const int* __restrict__ cumc,
                               const int* __restrict__ cumf,
                               float* __restrict__ cei, float* __restrict__ fei) {
  int e = blockIdx.x * blockDim.x + threadIdx.x;
  if (e >= N_EDGES) return;
  int r = ei[e];
  int c = ei[N_EDGES + e];
  bool m = maskf[e] > 0.5f;
  cei[e]           = m ? (float)(cumc[r] - 1) : -1.0f;
  cei[N_EDGES + e] = m ? (float)(cumc[c] - 1) : -1.0f;
  fei[e]           = !m ? (float)(cumf[r] - 1) : -1.0f;
  fei[N_EDGES + e] = !m ? (float)(cumf[c] - 1) : -1.0f;
}

// ---------------------------------------------------------------------------
extern "C" void kernel_launch(void* const* d_in, const int* in_sizes, int n_in,
                              void* d_out, int out_size, void* d_ws, size_t ws_size,
                              hipStream_t stream) {
  const float* x     = (const float*)d_in[0];
  const int*   ei    = (const int*)d_in[1];
  const int*   batch = (const int*)d_in[2];
  const float* W     = (const float*)d_in[3];
  const float* b     = (const float*)d_in[4];

  // ---- output layout (reference tuple, flattened in return order, f32) ----
  float* o = (float*)d_out;
  const size_t NND = (size_t)N_NODES * DIM;
  float* causal_x = o;                       o += NND;
  float* causal_ei = o;                      o += 2 * (size_t)N_EDGES;
  float* causal_w = o;                       o += N_EDGES;
  float* causal_b = o;                       o += N_NODES;
  float* n_causal = o;                       o += 1;
  float* conf_x = o;                         o += NND;
  float* conf_ei = o;                        o += 2 * (size_t)N_EDGES;
  float* conf_w = o;                         o += N_EDGES;
  float* conf_b = o;                         o += N_NODES;
  float* n_conf = o;                         o += 1;
  float* edge_score = o;                     o += N_EDGES;
  float* maskf = o;

  // ---- workspace layout (~2.5 MB) ----
  char* w = (char*)d_ws;
  float*    s_row  = (float*)w;    w += (size_t)N_NODES * 4;
  float*    s_col  = (float*)w;    w += (size_t)N_NODES * 4;
  unsigned* thr    = (unsigned*)w; w += (size_t)N_GRAPHS * 4;
  int*      quota  = (int*)w;      w += (size_t)N_GRAPHS * 4;
  int*      tiecnt = (int*)w;      w += (size_t)N_GRAPHS * 4;
  int*      selc   = (int*)w;      w += (size_t)N_NODES * 4;
  int*      self_  = (int*)w;      w += (size_t)N_NODES * 4;
  int*      cumc   = (int*)w;      w += (size_t)N_NODES * 4;
  int*      cumf   = (int*)w;      w += (size_t)N_NODES * 4;
  int*      bsums  = (int*)w;      w += (size_t)(2 * SCAN_BLOCKS) * 4;
  int*      totals = (int*)w;      w += 2 * 4;

  const int eblocks = (N_EDGES + 255) / 256;
  const int nblocks = (N_NODES + 255) / 256;

  score_wmma_kernel<<<N_NODES / 16, 32, 0, stream>>>(x, W, s_row, s_col);
  edge_score_kernel<<<eblocks, 256, 0, stream>>>(s_row, s_col, ei, b, edge_score);
  init_kernel<<<nblocks, 256, 0, stream>>>(selc, self_, tiecnt);
  topk_thresh_kernel<<<N_GRAPHS, 256, 0, stream>>>(edge_score, thr, quota);
  mask_kernel<<<eblocks, 256, 0, stream>>>(edge_score, thr, quota, tiecnt, ei,
                                           selc, self_, maskf, causal_w, conf_w);
  scan1_kernel<<<SCAN_BLOCKS, SCAN_T, 0, stream>>>(selc, self_, cumc, cumf, bsums);
  scan2_kernel<<<1, 128, 0, stream>>>(bsums, totals, n_causal, n_conf);
  scan3_kernel<<<nblocks, 256, 0, stream>>>(cumc, cumf, bsums);
  scatter_x_kernel<<<N_NODES, 64, 0, stream>>>(x, batch, selc, self_, cumc, cumf,
                                               totals, causal_x, causal_b,
                                               conf_x, conf_b);
  relabel_kernel<<<eblocks, 256, 0, stream>>>(ei, maskf, cumc, cumf,
                                              causal_ei, conf_ei);
}
